// DIHN_Loss_5720896438486
// MI455X (gfx1250) — matrix operand order repaired and checked by reading, not compiled, not run on hardware
//
#include <hip/hip_runtime.h>
#include <hip/hip_bf16.h>

// ---------------------------------------------------------------------------
// DIHN loss on MI455X (gfx1250, wave32, WMMA).
//
//   hash  = || 64*S - F @ B^T ||_F^2        (F:5000x64, B:20000x64, S:5000x20000)
//   quant = || F - B[omega] ||^2
//   corr  = sum(F)
//   loss  = (hash + 200*quant + 50*corr) / (n*m)
//
// Roofline: GEMM = 12.8 GFLOP; S stream = 400 MB -> ~17 us @ 23.3 TB/s.
// F/B (6.4 MB as bf16 hi/lo) are L2-resident, so we fuse GEMM + residual +
// reduction (the n x m intermediate never hits memory) on the bf16 WMMA path
// with a 3-term bf16 hi/lo split for ~fp32 accuracy:
//   A*B ~= Ahi*Bhi + Ahi*Blo + Alo*Bhi   (f32 accumulation)
// Main kernel: only the 312 FULL row-tiles (rows 0..4991): 78000 wave-tasks,
// exactly 8 waves/block -> no tail, no clamping, no divergence; 30 WMMAs and
// 40 imm-offset S loads per task. The ragged 8 rows (4992..4999, 0.16% of the
// GEMM) go to a tiny fp32 dot-product kernel.
// ---------------------------------------------------------------------------

typedef __attribute__((ext_vector_type(16))) __bf16 v16bf;
typedef __attribute__((ext_vector_type(8)))  __bf16 v8bf;
typedef __attribute__((ext_vector_type(8)))  float  v8f;

#define N_ROWS   5000
#define M_ROWS   20000
#define LCODE    64
#define ROW_TILES_MAIN 312            // full 16-row tiles (rows 0..4991)
#define EDGE_ROW0 4992                // remaining 8 rows
#define EDGE_ROWS 8
#define COL_TILES 1250                // 20000 / 16
#define NT_PER_WAVE 5                 // N-tiles per wave (reuses A fragments)
#define COL_GROUPS  250               // COL_TILES / NT_PER_WAVE
#define NTASK (312L * 250L)           // 78000 wave tasks, % 8 == 0
#define WAVES_PER_BLOCK 8

union V16U { v16bf v; v8bf h[2]; };

__device__ __forceinline__ v8f wmma_bf16(v16bf a, v16bf b, v8f c) {
  // 8 args: (neg_a, A, neg_b, B, c_mod, C, reuse_a, reuse_b)
  return __builtin_amdgcn_wmma_f32_16x16x32_bf16(false, a, false, b,
                                                 (short)0, c, false, false);
}

// ---- zero the 3 accumulators ------------------------------------------------
__global__ void zero_acc_kernel(float* acc) {
  if (threadIdx.x < 3) acc[threadIdx.x] = 0.0f;
}

// ---- f32 -> (bf16 hi, bf16 lo) split ----------------------------------------
__global__ void split_bf16_kernel(const float* __restrict__ src,
                                  __bf16* __restrict__ hi,
                                  __bf16* __restrict__ lo,
                                  long total) {
  long idx = (long)blockIdx.x * blockDim.x + threadIdx.x;
  if (idx >= total) return;
  float v = src[idx];
  __bf16 h = (__bf16)v;                    // RNE round to bf16
  __bf16 l = (__bf16)(v - (float)h);       // residual
  hi[idx] = h;
  lo[idx] = l;
}

// ---- fused GEMM (F @ B^T) + residual vs 64*S + reduction (full tiles) -------
__global__ void __launch_bounds__(256)
gemm_resid_kernel(const __bf16* __restrict__ Fh, const __bf16* __restrict__ Fl,
                  const __bf16* __restrict__ Bh, const __bf16* __restrict__ Bl,
                  const float*  __restrict__ S,  float* __restrict__ acc_hash) {
  const int wave = threadIdx.x >> 5;
  const int lane = threadIdx.x & 31;
  const int c = lane & 15;       // row (A) / col (B,C) within tile
  const int hh = lane >> 4;      // lane-half select

  const long task = (long)blockIdx.x * WAVES_PER_BLOCK + wave;  // < 78000 always
  const int ti   = (int)(task / COL_GROUPS);
  const int tg   = (int)(task % COL_GROUPS);
  const int row0 = ti * 16;                       // rows row0..row0+15 all valid
  const int colbase = tg * (NT_PER_WAVE * 16);    // 80 columns per task

  // ---- A fragments (16x32 bf16 per K-step), contiguous K order (same as B):
  //  lane l: row M = l%16, halves 0-7 at K = 32*s + 16*(l/16),
  //                        halves 8-15 at K = 32*s + 16*(l/16) + 8
  const __bf16* frow_h = Fh + (long)(row0 + c) * LCODE;
  const __bf16* frow_l = Fl + (long)(row0 + c) * LCODE;
  v16bf a_hi[2], a_lo[2];
#pragma unroll
  for (int s = 0; s < 2; ++s) {
    V16U ah, al;
    ah.h[0] = *(const v8bf*)(frow_h + 32 * s + 16 * hh);
    ah.h[1] = *(const v8bf*)(frow_h + 32 * s + 16 * hh + 8);
    al.h[0] = *(const v8bf*)(frow_l + 32 * s + 16 * hh);
    al.h[1] = *(const v8bf*)(frow_l + 32 * s + 16 * hh + 8);
    a_hi[s] = ah.v;
    a_lo[s] = al.v;
  }

  // Single base pointers; every per-(t,s,v) offset below is a compile-time
  // constant that fits the 24-bit global_load instruction offset.
  const __bf16* bbase_h = Bh + (long)(colbase + c) * LCODE;  // + t*1024 + 32s + 16hh
  const __bf16* bbase_l = Bl + (long)(colbase + c) * LCODE;
  const float*  sbase   = S + (long)(row0 + 8 * hh) * M_ROWS + colbase + c;  // + v*20000 + t*16

  float local = 0.0f;

#pragma unroll
  for (int t = 0; t < NT_PER_WAVE; ++t) {
    // ---- B fragments (32x16 bf16 per K-step), contiguous K order (same as A):
    //  lane l: col N = l%16, halves 0-15 at K = 32*s + 16*(l/16)
    v8f acc = {};
#pragma unroll
    for (int s = 0; s < 2; ++s) {
      v16bf bhv = *(const v16bf*)(bbase_h + t * (16 * LCODE) + 32 * s + 16 * hh);
      v16bf blv = *(const v16bf*)(bbase_l + t * (16 * LCODE) + 32 * s + 16 * hh);
      acc = wmma_bf16(a_hi[s], bhv, acc);   // hi*hi
      acc = wmma_bf16(a_hi[s], blv, acc);   // hi*lo
      acc = wmma_bf16(a_lo[s], bhv, acc);   // lo*hi
    }

    // ---- residual vs 64*S; C layout: lane l, vgpr v -> (row v+8*(l/16), col l%16)
#pragma unroll
    for (int v = 0; v < 8; ++v) {
      const float sv = sbase[(long)v * M_ROWS + t * 16];
      const float d = 64.0f * sv - acc[v];
      local += d * d;
    }
  }

  // wave32 butterfly reduction
#pragma unroll
  for (int off = 16; off > 0; off >>= 1)
    local += __shfl_xor(local, off, 32);

  __shared__ float red[WAVES_PER_BLOCK];
  if (lane == 0) red[wave] = local;
  __syncthreads();
  if (threadIdx.x == 0) {
    float s = 0.0f;
#pragma unroll
    for (int i = 0; i < WAVES_PER_BLOCK; ++i) s += red[i];
    atomicAdd(acc_hash, s);
  }
}

// ---- hash contributions of the ragged rows 4992..4999 (full fp32) -----------
// 8 x 20000 dot products over K=64: 20.5 MFLOP, 640 KB of S -- negligible.
// Launched with exactly 625 blocks x 256 threads == 160000 = 8*20000 threads.
__global__ void __launch_bounds__(256)
edge_hash_kernel(const float* __restrict__ F, const float* __restrict__ B,
                 const float* __restrict__ S, float* __restrict__ acc_hash) {
  const int idx = blockIdx.x * 256 + threadIdx.x;      // [0, 160000)
  const int r = EDGE_ROW0 + idx / M_ROWS;
  const int j = idx % M_ROWS;
  const float* frow = F + (long)r * LCODE;
  const float* brow = B + (long)j * LCODE;
  float dot = 0.0f;
#pragma unroll
  for (int k = 0; k < LCODE; ++k) dot += frow[k] * brow[k];
  const float d = 64.0f * S[(long)r * M_ROWS + j] - dot;
  float local = d * d;

#pragma unroll
  for (int off = 16; off > 0; off >>= 1)
    local += __shfl_xor(local, off, 32);

  __shared__ float red[8];
  const int wave = threadIdx.x >> 5, lane = threadIdx.x & 31;
  if (lane == 0) red[wave] = local;
  __syncthreads();
  if (threadIdx.x == 0) {
    float s = 0.0f;
#pragma unroll
    for (int i = 0; i < 8; ++i) s += red[i];
    atomicAdd(acc_hash, s);
  }
}

// ---- quantization loss (gather) + correlation loss (sum F) ------------------
__global__ void __launch_bounds__(256)
quant_corr_kernel(const float* __restrict__ F, const float* __restrict__ B,
                  const int* __restrict__ omega,
                  float* __restrict__ acc_q, float* __restrict__ acc_c) {
  const long idx = (long)blockIdx.x * 256 + threadIdx.x;
  float q = 0.0f, cc = 0.0f;
  if (idx < (long)N_ROWS * LCODE) {
    const int r = (int)(idx >> 6);
    const int k = (int)(idx & 63);
    const float f = F[idx];
    const long j = (long)omega[r];
    const float d = f - B[j * LCODE + k];
    q = d * d;
    cc = f;
  }
#pragma unroll
  for (int off = 16; off > 0; off >>= 1) {
    q  += __shfl_xor(q,  off, 32);
    cc += __shfl_xor(cc, off, 32);
  }
  __shared__ float rq[8], rc[8];
  const int wave = threadIdx.x >> 5, lane = threadIdx.x & 31;
  if (lane == 0) { rq[wave] = q; rc[wave] = cc; }
  __syncthreads();
  if (threadIdx.x == 0) {
    float sq = 0.0f, sc = 0.0f;
#pragma unroll
    for (int i = 0; i < 8; ++i) { sq += rq[i]; sc += rc[i]; }
    atomicAdd(acc_q, sq);
    atomicAdd(acc_c, sc);
  }
}

// ---- finalize ---------------------------------------------------------------
__global__ void finalize_kernel(const float* __restrict__ acc, float* __restrict__ out) {
  if (threadIdx.x == 0) {
    const float inv_nm = 1.0f / ((float)N_ROWS * (float)M_ROWS);
    out[0] = (acc[0] + 200.0f * acc[1] + 50.0f * acc[2]) * inv_nm;
  }
}

extern "C" void kernel_launch(void* const* d_in, const int* in_sizes, int n_in,
                              void* d_out, int out_size, void* d_ws, size_t ws_size,
                              hipStream_t stream) {
  const float* F     = (const float*)d_in[0];   // 5000 x 64
  const float* B     = (const float*)d_in[1];   // 20000 x 64
  const float* S     = (const float*)d_in[2];   // 5000 x 20000
  const int*   omega = (const int*)d_in[3];     // 5000
  float* out = (float*)d_out;

  // Workspace layout: [0,12): 3 f32 accumulators (hash, quant, corr)
  //                   +256: Fh | Fl | Bh | Bl  (bf16 hi/lo splits)
  char* ws = (char*)d_ws;
  float*  acc = (float*)ws;
  const long FELEMS = (long)N_ROWS * LCODE;   // 320000
  const long BELEMS = (long)M_ROWS * LCODE;   // 1280000
  __bf16* Fh = (__bf16*)(ws + 256);
  __bf16* Fl = Fh + FELEMS;
  __bf16* Bh = Fl + FELEMS;
  __bf16* Bl = Bh + BELEMS;

  zero_acc_kernel<<<1, 32, 0, stream>>>(acc);

  split_bf16_kernel<<<(int)((FELEMS + 255) / 256), 256, 0, stream>>>(F, Fh, Fl, FELEMS);
  split_bf16_kernel<<<(int)((BELEMS + 255) / 256), 256, 0, stream>>>(B, Bh, Bl, BELEMS);

  gemm_resid_kernel<<<(int)(NTASK / WAVES_PER_BLOCK), 256, 0, stream>>>(   // 9750 blocks
      Fh, Fl, Bh, Bl, S, acc + 0);

  edge_hash_kernel<<<(EDGE_ROWS * M_ROWS) / 256, 256, 0, stream>>>(F, B, S, acc + 0); // 625 blocks

  quant_corr_kernel<<<(N_ROWS * LCODE + 255) / 256, 256, 0, stream>>>(F, B, omega, acc + 1, acc + 2);

  finalize_kernel<<<1, 32, 0, stream>>>(acc, out);
}
